// MultiHeadSelfAttention_47184510714008
// MI455X (gfx1250) — matrix operand order repaired and checked by reading, not compiled, not run on hardware
//
#include <hip/hip_runtime.h>

typedef __attribute__((ext_vector_type(16))) _Float16 v16h;
typedef __attribute__((ext_vector_type(8)))  _Float16 v8h;
typedef __attribute__((ext_vector_type(4)))  _Float16 v4h;
typedef __attribute__((ext_vector_type(8)))  float    v8f;

#define BATCH   4
#define SEQ     2048
#define INCH    1024
#define NH      8
#define DHQK    64
#define DHV     128
// 1/sqrt(64) * log2(e): scores land in log2 domain -> v_exp_f32 used directly
#define QSCALE_LOG2E 0.18033688011112042f

__device__ __forceinline__ v16h mk16(v8h a, v8h b) {
  return __builtin_shufflevector(a, b, 0,1,2,3,4,5,6,7,8,9,10,11,12,13,14,15);
}

__device__ __forceinline__ v8f wmma_f16(v16h a, v16h b, v8f c) {
  return __builtin_amdgcn_wmma_f32_16x16x32_f16(
      false, a, false, b, (short)0, c, false, false);
}

// ---- gfx1250 async global->LDS copies (ASYNCcnt) -------------------------
__device__ __forceinline__ void async_b128(const void* g, void* l) {
  unsigned loff = (unsigned)(unsigned long long)
      (__attribute__((address_space(3))) const char*)l;
  unsigned long long ga = (unsigned long long)g;
  asm volatile("global_load_async_to_lds_b128 %0, %1, off"
               :: "v"(loff), "v"(ga) : "memory");
}
template <int N>
__device__ __forceinline__ void wait_async_le() {
  asm volatile("s_wait_asynccnt %0" :: "i"(N) : "memory");
}

// ---- 16-lane butterfly reductions in the VALU via DPP --------------------
template <int CTRL>
__device__ __forceinline__ float dpp_f(float x) {
  return __int_as_float(__builtin_amdgcn_mov_dpp(
      __float_as_int(x), CTRL, 0xf, 0xf, true));
}
__device__ __forceinline__ float red_max16(float v) {
  v = fmaxf(v, dpp_f<0xB1>(v));    // quad_perm(1,0,3,2)  : xor 1
  v = fmaxf(v, dpp_f<0x4E>(v));    // quad_perm(2,3,0,1)  : xor 2
  v = fmaxf(v, dpp_f<0x141>(v));   // row_half_mirror     : xor 7
  v = fmaxf(v, dpp_f<0x140>(v));   // row_mirror          : xor 15
  return v;
}
__device__ __forceinline__ float red_sum16(float v) {
  v += dpp_f<0xB1>(v);
  v += dpp_f<0x4E>(v);
  v += dpp_f<0x141>(v);
  v += dpp_f<0x140>(v);
  return v;
}

// ---------------------------------------------------------------------------
// One-time converts: x -> f16; W -> f16 transposed Wt[out][in].
// ---------------------------------------------------------------------------
__global__ __launch_bounds__(256)
void cvt_x_kernel(const float* __restrict__ X, _Float16* __restrict__ Xh) {
  size_t i = ((size_t)blockIdx.x * 256 + threadIdx.x) * 4;
  float4 f = *(const float4*)&X[i];
  v4h h; h[0] = (_Float16)f.x; h[1] = (_Float16)f.y;
         h[2] = (_Float16)f.z; h[3] = (_Float16)f.w;
  *(v4h*)&Xh[i] = h;
}

__global__ __launch_bounds__(256)
void cvt_wT_kernel(const float* __restrict__ W, _Float16* __restrict__ Wt, int OUTD) {
  size_t e = (size_t)blockIdx.x * 256 + threadIdx.x;   // linear into Wt[c][k]
  int k = (int)(e & (INCH - 1));
  int c = (int)(e >> 10);
  Wt[e] = (_Float16)W[(size_t)k * OUTD + c];
}

// ---------------------------------------------------------------------------
// Projection GEMM: 256 thr / 8 waves; tile 32 rows x 128 cols; K-step 64.
// Double-buffered async staging (5 async ops/thread/tile -> s_wait_asynccnt 5).
// Wave w owns cols [w*16,w*16+16) with two M accumulators (B-frag reuse).
// ---------------------------------------------------------------------------
template <int OUTD, int DH, bool VT, bool SCALEQ>
__global__ __launch_bounds__(256)
void qkv_gemm_kernel(const _Float16* __restrict__ Xh,
                     const _Float16* __restrict__ Wt,
                     const float* __restrict__ bias,
                     _Float16* __restrict__ Y) {
  __shared__ __align__(16) _Float16 Xs[2][32 * 64];    // [buf][row][k]  4KB ea
  __shared__ __align__(16) _Float16 Ws[2][128 * 64];   // [buf][col][k] 16KB ea

  const int tid  = threadIdx.x;
  const int lane = tid & 31;
  const int wv   = tid >> 5;
  const int hi   = (lane >> 4) & 1;
  const int ln   = lane & 15;
  const int r0   = blockIdx.x * 32;
  const int c0   = blockIdx.y * 128;

  // per-thread staging coordinates (uniform: 1 + 4 async ops per tile)
  const int xrow = tid >> 3, xkq = (tid & 7) * 8;

  auto stage = [&](int bi, int kk) {
    async_b128(&Xh[(size_t)(r0 + xrow) * INCH + kk + xkq],
               &Xs[bi][xrow * 64 + xkq]);
#pragma unroll
    for (int i = 0; i < 4; ++i) {
      int e  = i * 256 + tid;
      int c  = e >> 3;
      int kq = (e & 7) * 8;
      async_b128(&Wt[(size_t)(c0 + c) * INCH + kk + kq], &Ws[bi][c * 64 + kq]);
    }
  };

  v8f acc0 = {}, acc1 = {};
  stage(0, 0);

  constexpr int NIT = INCH / 64;   // 16
  for (int it = 0; it < NIT; ++it) {
    const int cur = it & 1;
    if (it + 1 < NIT) {
      stage(cur ^ 1, (it + 1) * 64);
      wait_async_le<5>();          // only tile `it` must be complete
    } else {
      wait_async_le<0>();
    }
    __syncthreads();

#pragma unroll
    for (int ks = 0; ks < 2; ++ks) {
      int ka = ks * 32 + (hi ? 8 : 0);
      v16h a0 = mk16(*(const v8h*)&Xs[cur][ln * 64 + ka],
                     *(const v8h*)&Xs[cur][ln * 64 + ka + 16]);
      v16h a1 = mk16(*(const v8h*)&Xs[cur][(16 + ln) * 64 + ka],
                     *(const v8h*)&Xs[cur][(16 + ln) * 64 + ka + 16]);
      int kb = ks * 32 + (hi ? 8 : 0);
      v16h b = mk16(*(const v8h*)&Ws[cur][(wv * 16 + ln) * 64 + kb],
                    *(const v8h*)&Ws[cur][(wv * 16 + ln) * 64 + kb + 16]);
      acc0 = wmma_f16(a0, b, acc0);
      acc1 = wmma_f16(a1, b, acc1);
    }
    __syncthreads();
  }

  const int col = c0 + wv * 16 + ln;
  const int h = col / DH;
  const int d = col % DH;
  const float bb = bias[col];
  constexpr float osc = SCALEQ ? QSCALE_LOG2E : 1.0f;
#pragma unroll
  for (int m = 0; m < 2; ++m) {
    const v8f& acc = m ? acc1 : acc0;
#pragma unroll
    for (int r = 0; r < 8; ++r) {
      int row  = r0 + m * 16 + r + (hi ? 8 : 0);
      int bidx = row >> 11;
      int n    = row & (SEQ - 1);
      float v = (acc[r] + bb) * osc;
      size_t idx = VT ? (((size_t)(bidx * NH + h) * DH + d) * SEQ + n)
                      : (((size_t)(bidx * NH + h) * SEQ + n) * DH + d);
      Y[idx] = (_Float16)v;
    }
  }
}

// ---------------------------------------------------------------------------
// Flash attention: block = 8 waves = one (b,h) x 128 queries; wave = 16 rows.
// Double-buffered async K/V staging (3 async ops/thread/tile).
// Softmax in exp2 domain; 16-lane reductions via DPP (no LDS traffic).
// ---------------------------------------------------------------------------
__global__ __launch_bounds__(256)
void flash_attn_kernel(const _Float16* __restrict__ Qh,   // [bh][n][64], pre-scaled
                       const _Float16* __restrict__ Kh,   // [bh][n][64]
                       const _Float16* __restrict__ Vt_g, // [bh][vcol][n]
                       float* __restrict__ Out) {
  __shared__ __align__(16) _Float16 Ks[2][32 * 64];      // [buf][key][d]   4KB ea
  __shared__ __align__(16) _Float16 Vt[2][128 * 32];     // [buf][vcol][key] 8KB ea
  __shared__ __align__(16) _Float16 Pb[8 * 16 * 32];     // per-wave P scratch

  const int tid  = threadIdx.x;
  const int lane = tid & 31;
  const int wv   = tid >> 5;
  const int hi   = (lane >> 4) & 1;
  const int ln   = lane & 15;
  const int bh   = blockIdx.y;
  const int q0   = blockIdx.x * 128 + wv * 16;

  auto stage = [&](int bi, int j0) {
    // K tile: 4KB contiguous -> 1 x b128 per thread
    async_b128(&Kh[((size_t)bh * SEQ + j0) * DHQK + tid * 8], &Ks[bi][tid * 8]);
    // Vt tile: 8KB, rows contiguous in Vt_g -> 2 x b128 per thread
#pragma unroll
    for (int i = 0; i < 2; ++i) {
      int e  = i * 256 + tid;
      int c  = e >> 2;
      int kq = (e & 3) * 8;
      async_b128(&Vt_g[((size_t)bh * DHV + c) * SEQ + j0 + kq],
                 &Vt[bi][c * 32 + kq]);
    }
  };

  // Q fragments (d = 0..63); Q already carries 1/sqrt(d)*log2(e)
  const _Float16* Qbase = Qh + ((size_t)bh * SEQ + q0) * DHQK;
  v16h qa[2];
#pragma unroll
  for (int ks = 0; ks < 2; ++ks) {
    int k0 = ks * 32 + (hi ? 8 : 0);
    qa[ks] = mk16(*(const v8h*)&Qbase[(size_t)ln * DHQK + k0],
                  *(const v8h*)&Qbase[(size_t)ln * DHQK + k0 + 16]);
  }

  v8f o[8];
  float m8[8], l8[8];
#pragma unroll
  for (int s = 0; s < 8; ++s) { v8f z = {}; o[s] = z; }
#pragma unroll
  for (int r = 0; r < 8; ++r) { m8[r] = -3.0e38f; l8[r] = 0.0f; }

  _Float16* Pw = &Pb[wv * 16 * 32];

  stage(0, 0);
  constexpr int NT = SEQ / 32;   // 64
  for (int jt = 0; jt < NT; ++jt) {
    const int cur = jt & 1;
    if (jt + 1 < NT) {
      stage(cur ^ 1, (jt + 1) * 32);
      wait_async_le<3>();
    } else {
      wait_async_le<0>();
    }
    __syncthreads();

    // S = Q @ K^T : two 16x16 tiles (log2-domain scores)
    v8f s0 = {}, s1 = {};
#pragma unroll
    for (int ks = 0; ks < 2; ++ks) {
      int kb = ks * 32 + (hi ? 8 : 0);
      s0 = wmma_f16(qa[ks], mk16(*(const v8h*)&Ks[cur][ln * 64 + kb], *(const v8h*)&Ks[cur][ln * 64 + kb + 16]), s0);
      s1 = wmma_f16(qa[ks], mk16(*(const v8h*)&Ks[cur][(16 + ln) * 64 + kb], *(const v8h*)&Ks[cur][(16 + ln) * 64 + kb + 16]), s1);
    }

    // online softmax, base-2
#pragma unroll
    for (int r = 0; r < 8; ++r) {
      float a0 = s0[r], a1 = s1[r];
      float mx   = red_max16(fmaxf(a0, a1));
      float mnew = fmaxf(m8[r], mx);
      float alpha = __builtin_amdgcn_exp2f(m8[r] - mnew);
      float p0 = __builtin_amdgcn_exp2f(a0 - mnew);
      float p1 = __builtin_amdgcn_exp2f(a1 - mnew);
      float rs = red_sum16(p0 + p1);
      l8[r] = l8[r] * alpha + rs;
      m8[r] = mnew;
#pragma unroll
      for (int s = 0; s < 8; ++s) o[s][r] *= alpha;
      int row = r + (hi ? 8 : 0);
      Pw[row * 32 + ln]      = (_Float16)p0;
      Pw[row * 32 + 16 + ln] = (_Float16)p1;
    }
    __builtin_amdgcn_wave_barrier();
    asm volatile("s_wait_dscnt 0x0" ::: "memory");  // wave-internal LDS RAW

    // O += P @ V
    {
      const int ka = hi ? 8 : 0;
      v16h pa = mk16(*(const v8h*)&Pw[ln * 32 + ka],
                     *(const v8h*)&Pw[ln * 32 + ka + 16]);
      const int kb = hi ? 8 : 0;
#pragma unroll
      for (int s = 0; s < 8; ++s) {
        v16h bvf = mk16(*(const v8h*)&Vt[cur][(s * 16 + ln) * 32 + kb],
                        *(const v8h*)&Vt[cur][(s * 16 + ln) * 32 + kb + 16]);
        o[s] = wmma_f16(pa, bvf, o[s]);
      }
    }
    __syncthreads();
  }

  // epilogue: out[b, n, h*128 + v]
  const int b = bh >> 3, h = bh & 7;
#pragma unroll
  for (int r = 0; r < 8; ++r) {
    float inv = __builtin_amdgcn_rcpf(l8[r]);
    int row = q0 + r + (hi ? 8 : 0);
#pragma unroll
    for (int s = 0; s < 8; ++s) {
      size_t idx = ((size_t)b * SEQ + row) * (NH * DHV) + h * DHV + s * 16 + ln;
      Out[idx] = o[s][r] * inv;
    }
  }
}

// ---------------------------------------------------------------------------
extern "C" void kernel_launch(void* const* d_in, const int* in_sizes, int n_in,
                              void* d_out, int out_size, void* d_ws, size_t ws_size,
                              hipStream_t stream) {
  (void)in_sizes; (void)n_in; (void)out_size; (void)ws_size;
  const float* x  = (const float*)d_in[0];
  const float* Wq = (const float*)d_in[1];
  const float* bq = (const float*)d_in[2];
  const float* Wk = (const float*)d_in[3];
  const float* bk = (const float*)d_in[4];
  const float* Wv = (const float*)d_in[5];
  const float* bv = (const float*)d_in[6];
  float* out = (float*)d_out;

  _Float16* Xh  = (_Float16*)d_ws;
  _Float16* Wqt = Xh  + (size_t)BATCH * SEQ * INCH;
  _Float16* Wkt = Wqt + (size_t)512 * INCH;
  _Float16* Wvt = Wkt + (size_t)512 * INCH;
  _Float16* Qh  = Wvt + (size_t)1024 * INCH;
  _Float16* Kh  = Qh  + (size_t)BATCH * NH * SEQ * DHQK;
  _Float16* Vtg = Kh  + (size_t)BATCH * NH * SEQ * DHQK;

  dim3 blk(256);
  cvt_x_kernel<<<dim3((BATCH * SEQ * INCH) / 1024), blk, 0, stream>>>(x, Xh);
  cvt_wT_kernel<<<dim3((512  * INCH) / 256), blk, 0, stream>>>(Wq, Wqt, 512);
  cvt_wT_kernel<<<dim3((512  * INCH) / 256), blk, 0, stream>>>(Wk, Wkt, 512);
  cvt_wT_kernel<<<dim3((1024 * INCH) / 256), blk, 0, stream>>>(Wv, Wvt, 1024);

  qkv_gemm_kernel<512,  DHQK, false, true ><<<dim3(BATCH * SEQ / 32, 4), blk, 0, stream>>>(Xh, Wqt, bq, Qh);
  qkv_gemm_kernel<512,  DHQK, false, false><<<dim3(BATCH * SEQ / 32, 4), blk, 0, stream>>>(Xh, Wkt, bk, Kh);
  qkv_gemm_kernel<1024, DHV,  true,  false><<<dim3(BATCH * SEQ / 32, 8), blk, 0, stream>>>(Xh, Wvt, bv, Vtg);

  flash_attn_kernel<<<dim3(SEQ / 128, BATCH * NH), blk, 0, stream>>>(Qh, Kh, Vtg, out);
}
